// BiAttention_54219667145018
// MI455X (gfx1250) — matrix operand order repaired and checked by reading, not compiled, not run on hardware
//
#include <hip/hip_runtime.h>

typedef __attribute__((ext_vector_type(16))) _Float16 v16h;
typedef __attribute__((ext_vector_type(8)))  _Float16 v8h;
typedef __attribute__((ext_vector_type(8)))  float    v8f;
typedef __attribute__((ext_vector_type(4)))  unsigned u32x4;
typedef __attribute__((ext_vector_type(8)))  int      i32x8;
typedef __attribute__((ext_vector_type(4)))  int      i32x4;

#define DEVFN static __device__ __forceinline__

constexpr int B_  = 32;
constexpr int N_  = 512;
constexpr int DM  = 512;   // d_model
constexpr int H_  = 8;
constexpr int DK  = 64;
constexpr int NB  = 9;     // N_B + 1 buckets
constexpr size_t QKV_ELEMS = (size_t)B_ * H_ * N_ * DK;       // 8,388,608 per tensor
constexpr size_t WH_ELEMS  = (size_t)6 * NB * DM * DM;        // f16 transposed weights

DEVFN v16h cat16(v8h lo, v8h hi) {
    return __builtin_shufflevector(lo, hi, 0,1,2,3,4,5,6,7,8,9,10,11,12,13,14,15);
}

DEVFN v8f wmma_f16(v16h a, v16h b, v8f c) {
    return __builtin_amdgcn_wmma_f32_16x16x32_f16(false, a, false, b, (short)0, c, false, false);
}

// ---------------------------------------------------------------------------
// Kernel 0: convert + transpose weights to f16:  Wh[t][c][col][d] = W[t%3][c][d][col]
// grid = (DM/32, DM/32, 54), block = 256.
// ---------------------------------------------------------------------------
__global__ __launch_bounds__(256)
void prep_w_kernel(const float* __restrict__ W_item,
                   const float* __restrict__ W_type,
                   _Float16* __restrict__ Wh)
{
    const int c0 = blockIdx.x * 32;     // col tile
    const int d0 = blockIdx.y * 32;     // d tile
    const int z  = blockIdx.z;          // t*NB + c,  t in 0..5
    const int t  = z / NB;
    const int c  = z % NB;

    const float* Wsrc = ((t < 3) ? W_item : W_type)
                      + ((size_t)(t % 3) * NB + c) * DM * DM;

    __shared__ _Float16 T[32][33];
    const int tid = threadIdx.x;
    const int col = tid & 31;
    #pragma unroll
    for (int rep = 0; rep < 4; ++rep) {
        int row = rep * 8 + (tid >> 5);
        T[row][col] = (_Float16)Wsrc[(size_t)(d0 + row) * DM + (c0 + col)];
    }
    __syncthreads();
    _Float16* dst = Wh + (size_t)z * DM * DM;
    #pragma unroll
    for (int rep = 0; rep < 4; ++rep) {
        int row = rep * 8 + (tid >> 5);
        dst[(size_t)(c0 + row) * DM + (d0 + col)] = T[col][row];
    }
}

// ---------------------------------------------------------------------------
// Kernel 1: group tokens by bucket, per batch.  1 block per batch, 512 thr.
// ---------------------------------------------------------------------------
__global__ __launch_bounds__(512)
void bucket_kernel(const int* __restrict__ b_seq,
                   int* __restrict__ g_counts,   // [B][NB]
                   int* __restrict__ g_idx)      // [B][NB][N]
{
    __shared__ int cnt[NB];
    int b = blockIdx.x;
    int n = threadIdx.x;
    if (n < NB) cnt[n] = 0;
    __syncthreads();
    int c = b_seq[(size_t)b * N_ + n];
    c = c < 0 ? 0 : (c > NB - 1 ? NB - 1 : c);
    int slot = atomicAdd(&cnt[c], 1);
    g_idx[((size_t)b * NB + c) * N_ + slot] = n;
    __syncthreads();
    if (n < NB) g_counts[b * NB + n] = cnt[n];
}

// ---------------------------------------------------------------------------
// Kernel 2: grouped projection GEMM with TDM-staged weights.
// grid = (32 token-tiles, B*NB, 6 tensors), block = 256 (8 waves).
// Per k-chunk the 512x32 f16 weight tile is DMA'd to LDS by the Tensor Data
// Mover (pad fields create the 80B row pitch fragments expect).
// ---------------------------------------------------------------------------
__global__ __launch_bounds__(256)
void proj_kernel(const float* __restrict__ hidden1,
                 const float* __restrict__ hidden2,
                 const _Float16* __restrict__ Wh,   // [6][NB][col 512][d 512] f16
                 const int*   __restrict__ g_counts,
                 const int*   __restrict__ g_idx,
                 _Float16*    __restrict__ qkv)
{
    const int tile = blockIdx.x;
    const int bc   = blockIdx.y;
    const int t    = blockIdx.z;
    const int b    = bc / NB;

    const int count = g_counts[bc];
    if (tile * 16 >= count) return;      // uniform early-exit

    const float*    x     = (t < 3) ? hidden1 : hidden2;
    const _Float16* Wbase = Wh + (size_t)(t * NB + (bc % NB)) * DM * DM;

    __shared__ int tok[16];
    __shared__ int vld[16];
    __shared__ __align__(16) _Float16 As[16][40];     // 16 rows x 32 K (80B pitch)
    __shared__ __align__(16) _Float16 Wt[512][40];    // 512 cols x 32 K (80B pitch)

    const int tid  = threadIdx.x;
    const int lane = tid & 31;
    const int wave = tid >> 5;
    const int mrow = lane & 15;
    const int kh   = lane >> 4;

    if (tid < 16) {
        int r  = tile * 16 + tid;
        int ok = (r < count);
        tok[tid] = g_idx[(size_t)bc * N_ + (ok ? r : tile * 16)];
        vld[tid] = ok;
    }
    __syncthreads();

    const unsigned ldsWt = (unsigned)(uintptr_t)&Wt[0][0];   // flat LDS addr[31:0]

    v8f acc[4] = {v8f{}, v8f{}, v8f{}, v8f{}};

    for (int k0 = 0; k0 < DM; k0 += 32) {
#if __has_builtin(__builtin_amdgcn_tensor_load_to_lds)
        // ---- TDM: DMA Wh[col 0..511][k0..k0+31] f16 -> LDS (80B row pitch) ----
        if (tid < 32) {   // one wave issues; EXEC is ignored by tensor ops
            uint64_t ga = (uint64_t)(uintptr_t)(Wbase + k0);
            u32x4 g0 = { 1u,                                   // count=1, user mode
                         ldsWt,                                // lds_addr
                         (unsigned)ga,                         // global_addr[31:0]
                         (unsigned)((ga >> 32) & 0x1FFFFFFu) | (2u << 30) }; // [56:32]|type=2
            i32x8 g1 = { (int)((1u << 16)      // data_size = 2 bytes
                             | (1u << 20)      // pad_enable
                             | (3u << 22)      // pad_interval: 16 DWORDs (one 64B row)
                             | (3u << 25)),    // pad_amount: 4 DWORDs (16B)
                         (int)(512u << 16),    // tensor_dim0 = 512 (d)
                         (int)(512u << 16),    // tensor_dim1 = 512 (cols)
                         (int)(32u << 16),     // tile_dim0 = 32
                         (int)512,             // tile_dim1 = 512, tile_dim2 = 0
                         (int)512,             // tensor_dim0_stride = 512
                         0, 0 };
            i32x4 gz = { 0, 0, 0, 0 };
#if defined(__clang_major__) && __clang_major__ >= 23
            i32x8 gz8 = { 0, 0, 0, 0, 0, 0, 0, 0 };
            __builtin_amdgcn_tensor_load_to_lds(g0, g1, gz, gz, gz8, 0);
#else
            __builtin_amdgcn_tensor_load_to_lds(g0, g1, gz, gz, 0);
#endif
        }
#else
        // ---- fallback: manual f16 copy of the weight tile ----
        #pragma unroll
        for (int rep = 0; rep < 2; ++rep) {
            int col = tid + rep * 256;
            const _Float16* wp = Wbase + (size_t)col * DM + k0;
            #pragma unroll
            for (int j = 0; j < 4; ++j)
                *(v8h*)&Wt[col][j * 8] = *(const v8h*)(wp + j * 8);
        }
#endif
        // ---- stage A: 16 gathered tokens x 32 K (f32 -> f16), overlaps TDM ----
        {
            int e = tid;
            #pragma unroll
            for (int rep = 0; rep < 2; ++rep, e += 256) {
                int r = e >> 5, kk = e & 31;
                As[r][kk] = (_Float16)x[((size_t)b * N_ + tok[r]) * DM + k0 + kk];
            }
        }
#if __has_builtin(__builtin_amdgcn_tensor_load_to_lds)
        if (tid < 32) __builtin_amdgcn_s_wait_tensorcnt(0);
#endif
        __syncthreads();

        // A fragment: element i -> K = (i<8 ? i : i+8) + 8*kh
        v8h alo = *(const v8h*)&As[mrow][kh * 8];
        v8h ahi = *(const v8h*)&As[mrow][16 + kh * 8];
        v16h a  = cat16(alo, ahi);

        // load all 4 B fragments, then issue WMMAs back-to-back
        v16h bfr[4];
        #pragma unroll
        for (int ct = 0; ct < 4; ++ct) {
            int col = (wave * 4 + ct) * 16 + mrow;
            v8h blo = *(const v8h*)&Wt[col][kh * 8];        // B: element i -> K = (i<8 ? i : i+8) + 8*kh
            v8h bhi = *(const v8h*)&Wt[col][16 + kh * 8];
            bfr[ct] = cat16(blo, bhi);
        }
        #pragma unroll
        for (int ct = 0; ct < 4; ++ct)
            acc[ct] = wmma_f16(a, bfr[ct], acc[ct]);

        __syncthreads();
    }

    // ---- scatter results:  Q/K [b][h][n][dk],  V transposed [b][h][dk][n] ----
    _Float16* out = qkv + (size_t)t * QKV_ELEMS;
    const bool isV = (t % 3) == 2;
    #pragma unroll
    for (int ct = 0; ct < 4; ++ct) {
        int col = (wave * 4 + ct) * 16 + mrow;
        int h   = col >> 6;
        int kd  = col & 63;
        #pragma unroll
        for (int v = 0; v < 8; ++v) {
            int r = v + kh * 8;                       // D layout: row = v + 8*half
            if (!vld[r]) continue;
            int n = tok[r];
            size_t addr = isV
                ? (((size_t)b * H_ + h) * DK + kd) * N_ + n
                : (((size_t)b * H_ + h) * N_ + n) * DK + kd;
            out[addr] = (_Float16)acc[ct][v];
        }
    }
}

// ---------------------------------------------------------------------------
// Kernel 3: flash attention.  grid = (N/64, B*H, 2), block = 128 (4 waves).
// z=0: h1 = attn(tQ, iK, iV);  z=1: h2 = attn(iQ, tK, tV).
// ---------------------------------------------------------------------------
__global__ __launch_bounds__(128)
void attn_kernel(const _Float16* __restrict__ qkv,
                 const unsigned char* __restrict__ mask,
                 float* __restrict__ out)
{
    const int qblk = blockIdx.x;
    const int bh   = blockIdx.y;
    const int z    = blockIdx.z;
    const int b    = bh / H_;
    const int h    = bh % H_;
    const int tid  = threadIdx.x;
    const int lane = tid & 31;
    const int wave = tid >> 5;
    const int mrow = lane & 15;
    const int kh   = lane >> 4;
    const int q0   = qblk * 64 + wave * 16;

    const _Float16* Q = qkv + (size_t)(z == 0 ? 3 : 0) * QKV_ELEMS;
    const _Float16* K = qkv + (size_t)(z == 0 ? 1 : 4) * QKV_ELEMS;
    const _Float16* V = qkv + (size_t)(z == 0 ? 2 : 5) * QKV_ELEMS;

    const _Float16* Qrow = Q + ((size_t)bh * N_ + q0 + mrow) * DK;
    v16h qf[2];
    #pragma unroll
    for (int f = 0; f < 2; ++f) {
        v8h lo = *(const v8h*)(Qrow + f * 32 + kh * 8);
        v8h hi = *(const v8h*)(Qrow + f * 32 + 16 + kh * 8);
        qf[f] = cat16(lo, hi);
    }

    bool mq[8];
    #pragma unroll
    for (int v = 0; v < 8; ++v)
        mq[v] = mask[(size_t)b * N_ + q0 + v + kh * 8] != 0;

    float mi[8], li[8];
    #pragma unroll
    for (int v = 0; v < 8; ++v) { mi[v] = -1e30f; li[v] = 0.f; }
    v8f o[4] = {v8f{}, v8f{}, v8f{}, v8f{}};

    __shared__ __align__(16) _Float16 Pw[4][16][40];

    for (int j0 = 0; j0 < N_; j0 += 32) {
        // ---- scores: two 16x16 tiles over 32 keys ----
        v8f s[2];
        #pragma unroll
        for (int tI = 0; tI < 2; ++tI) {
            const _Float16* Krow = K + ((size_t)bh * N_ + j0 + tI * 16 + mrow) * DK;
            v16h kb[2];
            #pragma unroll
            for (int f = 0; f < 2; ++f) {
                v8h lo = *(const v8h*)(Krow + f * 32 + kh * 8);
                v8h hi = *(const v8h*)(Krow + f * 32 + 16 + kh * 8);
                kb[f] = cat16(lo, hi);
            }
            v8f a = v8f{};
            a = wmma_f16(qf[0], kb[0], a);
            a = wmma_f16(qf[1], kb[1], a);
            s[tI] = a;
        }

        const bool mp0 = mask[(size_t)b * N_ + j0 + mrow]      != 0;
        const bool mp1 = mask[(size_t)b * N_ + j0 + 16 + mrow] != 0;

        // ---- online softmax over the 32 new columns ----
        #pragma unroll
        for (int v = 0; v < 8; ++v) {
            float s0 = (mq[v] && mp0) ? s[0][v] * 0.125f : -1e30f;
            float s1 = (mq[v] && mp1) ? s[1][v] * 0.125f : -1e30f;
            float mx = fmaxf(s0, s1);
            #pragma unroll
            for (int d = 1; d < 16; d <<= 1)
                mx = fmaxf(mx, __shfl_xor(mx, d, 16));
            float mnew = fmaxf(mi[v], mx);
            float scl  = __expf(mi[v] - mnew);
            float p0   = __expf(s0 - mnew);
            float p1   = __expf(s1 - mnew);
            float rs   = p0 + p1;
            #pragma unroll
            for (int d = 1; d < 16; d <<= 1)
                rs += __shfl_xor(rs, d, 16);
            li[v] = li[v] * scl + rs;
            mi[v] = mnew;
            #pragma unroll
            for (int ct = 0; ct < 4; ++ct) o[ct][v] *= scl;
            int r = v + kh * 8;
            Pw[wave][r][mrow]      = (_Float16)p0;
            Pw[wave][r][16 + mrow] = (_Float16)p1;
        }
        asm volatile("s_wait_dscnt 0" ::: "memory");   // wave-local LDS W->R

        // ---- O += P * V ----
        v8h plo = *(const v8h*)&Pw[wave][mrow][kh * 8];
        v8h phi = *(const v8h*)&Pw[wave][mrow][16 + kh * 8];
        v16h pf = cat16(plo, phi);
        v16h vb[4];
        #pragma unroll
        for (int ct = 0; ct < 4; ++ct) {
            const _Float16* Vp = V + ((size_t)bh * DK + ct * 16 + mrow) * N_
                                   + j0 + kh * 8;
            vb[ct] = cat16(*(const v8h*)(Vp), *(const v8h*)(Vp + 16));
        }
        #pragma unroll
        for (int ct = 0; ct < 4; ++ct)
            o[ct] = wmma_f16(pf, vb[ct], o[ct]);
    }

    // ---- normalize and store [z][b][n][h*64+dk] f32 ----
    float* op = out + (size_t)z * B_ * N_ * DM;
    #pragma unroll
    for (int ct = 0; ct < 4; ++ct) {
        #pragma unroll
        for (int v = 0; v < 8; ++v) {
            int r = v + kh * 8;
            float l = li[v];
            float val = (l > 0.f) ? o[ct][v] / l : 0.f;
            op[((size_t)b * N_ + q0 + r) * DM + h * DK + ct * 16 + mrow] = val;
        }
    }
}

// ---------------------------------------------------------------------------
extern "C" void kernel_launch(void* const* d_in, const int* in_sizes, int n_in,
                              void* d_out, int out_size, void* d_ws, size_t ws_size,
                              hipStream_t stream) {
    const float*         hidden1 = (const float*)d_in[0];
    const float*         hidden2 = (const float*)d_in[1];
    const unsigned char* mask    = (const unsigned char*)d_in[2];
    const int*           b_seq   = (const int*)d_in[3];
    const float*         W_item  = (const float*)d_in[4];
    const float*         W_type  = (const float*)d_in[5];
    float*               out     = (float*)d_out;

    char* wsb = (char*)d_ws;
    _Float16* qkv = (_Float16*)wsb;                                      // 6 f16 tensors
    _Float16* Wh  = (_Float16*)(wsb + 6 * QKV_ELEMS * sizeof(_Float16)); // f16 W^T
    int* g_counts = (int*)(wsb + (6 * QKV_ELEMS + WH_ELEMS) * sizeof(_Float16));
    int* g_idx    = g_counts + B_ * NB;

    prep_w_kernel<<<dim3(DM / 32, DM / 32, 6 * NB), dim3(256), 0, stream>>>(
        W_item, W_type, Wh);

    bucket_kernel<<<dim3(B_), dim3(N_), 0, stream>>>(b_seq, g_counts, g_idx);

    dim3 pgrid(N_ / 16, B_ * NB, 6);
    proj_kernel<<<pgrid, dim3(256), 0, stream>>>(hidden1, hidden2, Wh,
                                                 g_counts, g_idx, qkv);

    dim3 agrid(N_ / 64, B_ * H_, 2);
    attn_kernel<<<agrid, dim3(128), 0, stream>>>(qkv, mask, out);
}
